// SparseAttention_42872363549272
// MI455X (gfx1250) — matrix operand order repaired and checked by reading, not compiled, not run on hardware
//
#include <hip/hip_runtime.h>
#include <hip/hip_bf16.h>
#include <math.h>

typedef __attribute__((ext_vector_type(16))) _Float16 v16h;
typedef __attribute__((ext_vector_type(8)))  float    v8f;

#define D_MODEL 1024
#define SEQ     2048
#define BATCH   2
#define HEADS   16
#define DH      64
#define TOPK    64

// ---------------- CDNA5 async global->LDS helpers ----------------

__device__ __forceinline__ void async_copy_b128(unsigned lds_byte_off,
                                                const void* gaddr) {
  // GLOBAL_LOAD_ASYNC_TO_LDS_B128: VDST = per-lane LDS byte address,
  // VADDR = 64-bit global address (GV mode). Tracked by ASYNCcnt.
  asm volatile("global_load_async_to_lds_b128 %0, %1, off"
               :
               : "v"(lds_byte_off), "v"((unsigned long long)(uintptr_t)gaddr)
               : "memory");
}

__device__ __forceinline__ unsigned lds_addr(const void* p) {
  // Generic pointers to LDS carry the LDS byte offset in addr[31:0].
  return (unsigned)(uintptr_t)p;
}

// ---------------- conversion kernels ----------------

__global__ void cvt_f16_kernel(const float* __restrict__ src,
                               _Float16* __restrict__ dst, int nElem) {
  int i = blockIdx.x * blockDim.x + threadIdx.x;
  int stride = gridDim.x * blockDim.x;
  for (; i < nElem; i += stride) dst[i] = (_Float16)src[i];
}

// W [K][N] f32 -> WT [N][K] f16, tiled 32x32 transpose
__global__ __launch_bounds__(256)
void cvt_transpose_kernel(const float* __restrict__ W, _Float16* __restrict__ WT) {
  __shared__ float tile[32][33];
  const int tx = threadIdx.x & 31;
  const int ty = threadIdx.x >> 5;            // 0..7
  const int kbase = blockIdx.y * 32;
  const int nbase = blockIdx.x * 32;
#pragma unroll
  for (int i = 0; i < 4; ++i)
    tile[ty + i * 8][tx] = W[(size_t)(kbase + ty + i * 8) * D_MODEL + nbase + tx];
  __syncthreads();
#pragma unroll
  for (int i = 0; i < 4; ++i)
    WT[(size_t)(nbase + ty + i * 8) * D_MODEL + kbase + tx] =
        (_Float16)tile[tx][ty + i * 8];
}

// ---------------- WMMA GEMM with async double-buffered LDS tiles ----------------
// Y[M][1024] = A[M][1024](f16) @ W + bias ; BT = W^T stored [n][k] f16.
// Block tile 128x128, BK=32, 8 waves; each wave computes 32(M) x 64(N)
// via a 2x4 grid of 16x16x32 WMMA tiles.  Tiles staged to LDS with
// GLOBAL_LOAD_ASYNC_TO_LDS_B128 (ASYNCcnt), double buffered, k-loop
// unrolled by 2 so each phase uses a compile-time-constant buffer.

template <bool OUT_F16>
__global__ __launch_bounds__(256)
void gemm_kernel(const _Float16* __restrict__ A,
                 const _Float16* __restrict__ BT,
                 const float* __restrict__ bias,
                 void* __restrict__ outv) {
  __shared__ __align__(16) _Float16 Ash[2][128 * 40];
  __shared__ __align__(16) _Float16 Bsh[2][128 * 40];

  const int tid  = threadIdx.x;
  const int lane = tid & 31;
  const int w    = tid >> 5;      // 0..7
  const int wr   = w >> 1;        // 0..3  (M)
  const int wc   = w & 1;         // 0..1  (N)
  const int l16  = lane & 15;
  const int hi   = lane >> 4;     // 0 or 1
  const int rowBase = blockIdx.y * 128;
  const int colBase = blockIdx.x * 128;
  const int KCH = D_MODEL / 32;   // 32 k-chunks (even)

  // this thread's two (row, kgroup) slots within the 128x32 tile
  const int r0 = tid >> 2,          kg0 = tid & 3;
  const int r1 = (tid + 256) >> 2,  kg1 = (tid + 256) & 3;

  // per-thread global row bases (k offset added per chunk)
  const _Float16* gA0 = A  + (size_t)(rowBase + r0) * D_MODEL + kg0 * 8;
  const _Float16* gA1 = A  + (size_t)(rowBase + r1) * D_MODEL + kg1 * 8;
  const _Float16* gB0 = BT + (size_t)(colBase + r0) * D_MODEL + kg0 * 8;
  const _Float16* gB1 = BT + (size_t)(colBase + r1) * D_MODEL + kg1 * 8;

  // per-thread LDS destinations for each buffer
  const unsigned lA0[2] = { lds_addr(&Ash[0][r0 * 40 + kg0 * 8]),
                            lds_addr(&Ash[1][r0 * 40 + kg0 * 8]) };
  const unsigned lA1[2] = { lds_addr(&Ash[0][r1 * 40 + kg1 * 8]),
                            lds_addr(&Ash[1][r1 * 40 + kg1 * 8]) };
  const unsigned lB0[2] = { lds_addr(&Bsh[0][r0 * 40 + kg0 * 8]),
                            lds_addr(&Bsh[1][r0 * 40 + kg0 * 8]) };
  const unsigned lB1[2] = { lds_addr(&Bsh[0][r1 * 40 + kg1 * 8]),
                            lds_addr(&Bsh[1][r1 * 40 + kg1 * 8]) };

  v8f acc[2][4] = {};

  auto issue = [&](int chunk, int buf) {
    const int ko = chunk * 32;
    async_copy_b128(lA0[buf], gA0 + ko);
    async_copy_b128(lA1[buf], gA1 + ko);
    async_copy_b128(lB0[buf], gB0 + ko);
    async_copy_b128(lB1[buf], gB1 + ko);
  };

  auto compute = [&](int buf) {
    v16h af[2], bf[4];
#pragma unroll
    for (int mt = 0; mt < 2; ++mt) {
      // A 16x32 layout: lanes<16 hold K{0..7,16..23}; lanes>=16 K{8..15,24..31}
      const _Float16* ap = &Ash[buf][(wr * 32 + mt * 16 + l16) * 40];
      ((uint4*)&af[mt])[0] = *(const uint4*)(ap + hi * 8);
      ((uint4*)&af[mt])[1] = *(const uint4*)(ap + 16 + hi * 8);
    }
#pragma unroll
    for (int nt = 0; nt < 4; ++nt) {
      // B 32x16 layout: lanes<16 hold K{0..7,16..23} of column l16; lanes>=16 K{8..15,24..31}
      const _Float16* bp = &Bsh[buf][(wc * 64 + nt * 16 + l16) * 40];
      ((uint4*)&bf[nt])[0] = *(const uint4*)(bp + hi * 8);
      ((uint4*)&bf[nt])[1] = *(const uint4*)(bp + 16 + hi * 8);
    }
#pragma unroll
    for (int mt = 0; mt < 2; ++mt)
#pragma unroll
      for (int nt = 0; nt < 4; ++nt)
        acc[mt][nt] = __builtin_amdgcn_wmma_f32_16x16x32_f16(
            false, af[mt], false, bf[nt], (short)0, acc[mt][nt], false, false);
  };

  // prologue: two chunks in flight (8 async b128 per thread)
  issue(0, 0);
  issue(1, 1);

  for (int kc = 0; kc < KCH; kc += 2) {
    // ---- phase 0: chunk kc in buffer 0 ----
    // chunks kc and kc+1 outstanding (8); <=4 means chunk kc landed
    asm volatile("s_wait_asynccnt 0x4" ::: "memory");
    __syncthreads();
    compute(0);
    __syncthreads();               // everyone done reading buf0
    if (kc + 2 < KCH) {
      issue(kc + 2, 0);            // refill buf0
      asm volatile("s_wait_asynccnt 0x4" ::: "memory");  // chunk kc+1 landed
    } else {
      asm volatile("s_wait_asynccnt 0x0" ::: "memory");  // drain: kc+1 landed
    }
    // ---- phase 1: chunk kc+1 in buffer 1 ----
    __syncthreads();
    compute(1);
    __syncthreads();               // everyone done reading buf1
    if (kc + 3 < KCH) issue(kc + 3, 1);
  }

  // epilogue: C/D layout: VGPR j -> M = j + 8*hi ; N = l16
#pragma unroll
  for (int mt = 0; mt < 2; ++mt) {
#pragma unroll
    for (int nt = 0; nt < 4; ++nt) {
      int col = colBase + wc * 64 + nt * 16 + l16;
      float bsv = bias[col];
#pragma unroll
      for (int j = 0; j < 8; ++j) {
        int row = rowBase + wr * 32 + mt * 16 + hi * 8 + j;
        float val = acc[mt][nt][j] + bsv;
        if (OUT_F16)
          ((_Float16*)outv)[(size_t)row * D_MODEL + col] = (_Float16)val;
        else
          ((float*)outv)[(size_t)row * D_MODEL + col] = val;
      }
    }
  }
}

// ---------------- scores + top-k + softmax + context ----------------
// grid: (SEQ/16, HEADS, BATCH), 256 threads (8 waves).
// Dynamic LDS: sc[16][2048] f32 (128KB) + tw[16][64] + ti[16][64].

__global__ __launch_bounds__(256)
void attn_kernel(const _Float16* __restrict__ qh,
                 const _Float16* __restrict__ kh,
                 const float* __restrict__ vf,
                 _Float16* __restrict__ ctxh,
                 float* __restrict__ aw_out) {
  extern __shared__ __align__(16) char smem_raw[];
  float* sc = (float*)smem_raw;          // [16][2048]
  float* tw = sc + 16 * 2048;            // [16][64]
  int*   ti = (int*)(tw + 16 * 64);      // [16][64]

  const int qt   = blockIdx.x;           // query tile 0..127
  const int h    = blockIdx.y;
  const int b    = blockIdx.z;
  const int tid  = threadIdx.x;
  const int lane = tid & 31;
  const int w    = tid >> 5;
  const int l16  = lane & 15;
  const int hi   = lane >> 4;

  // Q A-fragments (16 queries x 64 dh = two K=32 chunks), loaded from global
  const _Float16* qrow =
      qh + ((size_t)(b * SEQ + qt * 16 + l16)) * D_MODEL + h * DH;
  v16h aq[2];
#pragma unroll
  for (int c = 0; c < 2; ++c) {
    ((uint4*)&aq[c])[0] = *(const uint4*)(qrow + c * 32 + hi * 8);
    ((uint4*)&aq[c])[1] = *(const uint4*)(qrow + c * 32 + 16 + hi * 8);
  }

  // scores: wave w covers keys [w*256, w*256+256)
  const float scale = 0.125f;  // 1/sqrt(dh)
  for (int kt = 0; kt < 16; ++kt) {
    int key0 = w * 256 + kt * 16;
    // B fragment: B[kdim][key] = K[key][kdim] -> contiguous in kh row-major
    const _Float16* krow =
        kh + ((size_t)(b * SEQ + key0 + l16)) * D_MODEL + h * DH;
    v16h bf0, bf1;
    ((uint4*)&bf0)[0] = *(const uint4*)(krow + hi * 8);
    ((uint4*)&bf0)[1] = *(const uint4*)(krow + 16 + hi * 8);
    ((uint4*)&bf1)[0] = *(const uint4*)(krow + 32 + hi * 8);
    ((uint4*)&bf1)[1] = *(const uint4*)(krow + 48 + hi * 8);
    v8f acc = {};
    acc = __builtin_amdgcn_wmma_f32_16x16x32_f16(false, aq[0], false, bf0,
                                                 (short)0, acc, false, false);
    acc = __builtin_amdgcn_wmma_f32_16x16x32_f16(false, aq[1], false, bf1,
                                                 (short)0, acc, false, false);
#pragma unroll
    for (int j = 0; j < 8; ++j)
      sc[(hi * 8 + j) * 2048 + key0 + l16] = acc[j] * scale;
  }
  __syncthreads();

  // each wave owns query rows 2w, 2w+1
  for (int rr = 0; rr < 2; ++rr) {
    int r = w * 2 + rr;
    float* srow = sc + r * 2048;

    // iterative top-64 (descending score, smaller index on ties, like lax.top_k)
    for (int sel = 0; sel < TOPK; ++sel) {
      float best = -INFINITY;
      int bidx = 0x7fffffff;
#pragma unroll 8
      for (int t = 0; t < 64; ++t) {
        int idx = lane + (t << 5);
        float s = srow[idx];
        if (s > best || (s == best && idx < bidx)) { best = s; bidx = idx; }
      }
#pragma unroll
      for (int off = 16; off > 0; off >>= 1) {
        float os = __shfl_xor(best, off, 32);
        int   oi = __shfl_xor(bidx, off, 32);
        if (os > best || (os == best && oi < bidx)) { best = os; bidx = oi; }
      }
      if ((bidx & 31) == lane) srow[bidx] = -INFINITY;  // mark taken
      if (lane == 0) { tw[r * 64 + sel] = best; ti[r * 64 + sel] = bidx; }
    }

    // softmax over the 64 selected (tw[r][0] is the max)
    float m  = tw[r * 64];
    float e0 = __expf(tw[r * 64 + lane] - m);
    float e1 = __expf(tw[r * 64 + lane + 32] - m);
    float ssum = e0 + e1;
#pragma unroll
    for (int off = 16; off > 0; off >>= 1) ssum += __shfl_xor(ssum, off, 32);
    float inv = 1.0f / ssum;
    float w0 = e0 * inv, w1 = e1 * inv;
    tw[r * 64 + lane] = w0;
    tw[r * 64 + lane + 32] = w1;
    size_t awBase = ((size_t)((b * HEADS + h) * SEQ + qt * 16 + r)) * TOPK;
    aw_out[awBase + lane] = w0;
    aw_out[awBase + lane + 32] = w1;

    // context: gather-weighted sum of V rows (each lane: dims lane, lane+32)
    float acc0 = 0.f, acc1 = 0.f;
    const float* vb = vf + ((size_t)b * SEQ) * D_MODEL + h * DH;
    for (int kk = 0; kk < TOPK; ++kk) {
      int idx = ti[r * 64 + kk];
      float wgt = tw[r * 64 + kk];
      const float* vp = vb + (size_t)idx * D_MODEL;
      acc0 += wgt * vp[lane];
      acc1 += wgt * vp[lane + 32];
    }
    size_t crow = ((size_t)(b * SEQ + qt * 16 + r)) * D_MODEL + h * DH;
    ctxh[crow + lane]      = (_Float16)acc0;
    ctxh[crow + lane + 32] = (_Float16)acc1;
  }
}

// ---------------- launch ----------------

extern "C" void kernel_launch(void* const* d_in, const int* in_sizes, int n_in,
                              void* d_out, int out_size, void* d_ws, size_t ws_size,
                              hipStream_t stream) {
  (void)in_sizes; (void)n_in; (void)out_size; (void)ws_size;
  const float* x  = (const float*)d_in[0];
  const float* Wq = (const float*)d_in[1];
  const float* bq = (const float*)d_in[2];
  const float* Wk = (const float*)d_in[3];
  const float* bk = (const float*)d_in[4];
  const float* Wv = (const float*)d_in[5];
  const float* bv = (const float*)d_in[6];
  const float* Wo = (const float*)d_in[7];
  const float* bo = (const float*)d_in[8];

  float* out    = (float*)d_out;                               // [B,n,d]
  float* aw_out = out + (size_t)BATCH * SEQ * D_MODEL;         // [B,H,n,64]

  char* ws = (char*)d_ws;
  _Float16* xh   = (_Float16*)(ws);                         // 8 MB
  _Float16* WqT  = (_Float16*)(ws + (8ull  << 20));         // 2 MB
  _Float16* WkT  = (_Float16*)(ws + (10ull << 20));         // 2 MB
  _Float16* WvT  = (_Float16*)(ws + (12ull << 20));         // 2 MB
  _Float16* WoT  = (_Float16*)(ws + (14ull << 20));         // 2 MB
  _Float16* qh   = (_Float16*)(ws + (16ull << 20));         // 8 MB
  _Float16* kh   = (_Float16*)(ws + (24ull << 20));         // 8 MB
  float*    vf   = (float*)   (ws + (32ull << 20));         // 16 MB
  _Float16* ctxh = (_Float16*)(ws + (48ull << 20));         // 8 MB

  const int M = BATCH * SEQ;  // 4096

  cvt_f16_kernel<<<1024, 256, 0, stream>>>(x, xh, M * D_MODEL);
  dim3 tg(D_MODEL / 32, D_MODEL / 32);
  cvt_transpose_kernel<<<tg, 256, 0, stream>>>(Wq, WqT);
  cvt_transpose_kernel<<<tg, 256, 0, stream>>>(Wk, WkT);
  cvt_transpose_kernel<<<tg, 256, 0, stream>>>(Wv, WvT);
  cvt_transpose_kernel<<<tg, 256, 0, stream>>>(Wo, WoT);

  dim3 gg(D_MODEL / 128, M / 128);  // (8, 32)
  gemm_kernel<true ><<<gg, 256, 0, stream>>>(xh, WqT, bq, (void*)qh);
  gemm_kernel<true ><<<gg, 256, 0, stream>>>(xh, WkT, bk, (void*)kh);
  gemm_kernel<false><<<gg, 256, 0, stream>>>(xh, WvT, bv, (void*)vf);

  size_t smem = (size_t)16 * 2048 * 4 + 16 * 64 * 4 + 16 * 64 * 4;  // ~136 KB
  attn_kernel<<<dim3(SEQ / 16, HEADS, BATCH), 256, smem, stream>>>(
      qh, kh, vf, ctxh, aw_out);

  gemm_kernel<false><<<gg, 256, 0, stream>>>(ctxh, WoT, bo, (void*)out);
}
